// SparseGatedMoE_64123861729766
// MI455X (gfx1250) — compile-verified, matrix operand order checked by reading, not run on hardware
//
#include <hip/hip_runtime.h>
#include <hip/hip_bf16.h>
#include <math.h>

#define D_ 1024
#define H_ 4096
#define E_ 8

typedef __attribute__((ext_vector_type(16))) __bf16 v16bf;
typedef __attribute__((ext_vector_type(8)))  float  v8f;

union Frag {
    v16bf  v;
    float4 f4[2];
};

// Load a 16x32 bf16 fragment (A or B layout, per CDNA5 ISA 7.12.2):
// lane holds row/col = lane&15, K = {klo..klo+7, 16+klo..16+klo+7}, klo=(lane>>4)*8.
// base must already point at (row_or_col * stride + k0 + klo).
__device__ __forceinline__ void load_frag(Frag& f, const __bf16* base) {
    f.f4[0] = *(const float4*)(base);       // K = k0+klo .. k0+klo+7
    f.f4[1] = *(const float4*)(base + 16);  // K = k0+16+klo .. k0+16+klo+7
}

// ---------------------------------------------------------------------------
// Gating: one wave (32 lanes) per token. Computes top-2 indices + softmax
// weights, and accumulates full-softmax importance for the aux loss.
// ---------------------------------------------------------------------------
__global__ __launch_bounds__(128) void gate_kernel(
    const float* __restrict__ x, const float* __restrict__ Wg,
    const float* __restrict__ bg, int* __restrict__ topi,
    float* __restrict__ topg, float* __restrict__ imp)
{
    int wave = threadIdx.x >> 5;
    int lane = threadIdx.x & 31;
    int n = blockIdx.x * 4 + wave;

    const float* xr = x + (size_t)n * D_;
    float acc[E_];
#pragma unroll
    for (int e = 0; e < E_; ++e) acc[e] = 0.f;

    for (int d = lane; d < D_; d += 32) {
        float xv = xr[d];
        const float* wr = Wg + d * E_;   // Wg is [D][E] row-major
#pragma unroll
        for (int e = 0; e < E_; ++e) acc[e] += xv * wr[e];
    }
#pragma unroll
    for (int e = 0; e < E_; ++e)
        for (int off = 16; off > 0; off >>= 1)
            acc[e] += __shfl_xor(acc[e], off, 32);

    if (lane == 0) {
        float lg[E_];
#pragma unroll
        for (int e = 0; e < E_; ++e) lg[e] = acc[e] + bg[e];
        // top-2 (ties -> lower index, matching lax.top_k)
        int i0 = 0;
        for (int e = 1; e < E_; ++e) if (lg[e] > lg[i0]) i0 = e;
        int i1 = (i0 == 0) ? 1 : 0;
        for (int e = 0; e < E_; ++e) if (e != i0 && lg[e] > lg[i1]) i1 = e;
        float m2 = lg[i0];
        float e0 = __expf(lg[i0] - m2), e1 = __expf(lg[i1] - m2);
        float inv = 1.f / (e0 + e1);
        topi[2 * n + 0] = i0;  topg[2 * n + 0] = e0 * inv;
        topi[2 * n + 1] = i1;  topg[2 * n + 1] = e1 * inv;
        // full softmax -> importance accumulation
        float mm = lg[0];
        for (int e = 1; e < E_; ++e) mm = fmaxf(mm, lg[e]);
        float ge[E_], s = 0.f;
#pragma unroll
        for (int e = 0; e < E_; ++e) { ge[e] = __expf(lg[e] - mm); s += ge[e]; }
        float is = 1.f / s;
#pragma unroll
        for (int e = 0; e < E_; ++e) atomicAdd(&imp[e], ge[e] * is);
    }
}

__global__ void zero_imp_kernel(float* imp) {
    if (threadIdx.x < E_) imp[threadIdx.x] = 0.f;
}

__global__ void aux_kernel(const float* __restrict__ imp, float* outp) {
    float s = 0.f;
    for (int e = 0; e < E_; ++e) s += imp[e];
    float mean = s / E_;
    float var = 0.f;
    for (int e = 0; e < E_; ++e) { float d = imp[e] - mean; var += d * d; }
    var /= (E_ - 1);                       // ddof=1
    float cv = sqrtf(var) / (mean + 1e-8f);
    outp[0] = cv * cv;
}

// ---------------------------------------------------------------------------
// fp32 -> bf16 elementwise
// ---------------------------------------------------------------------------
__global__ __launch_bounds__(256) void cvt_bf16_kernel(
    const float* __restrict__ in, __bf16* __restrict__ out, size_t n)
{
    size_t i = ((size_t)blockIdx.x * blockDim.x + threadIdx.x) * 4;
    if (i + 3 < n) {
        float4 v = *(const float4*)(in + i);
        out[i + 0] = (__bf16)v.x; out[i + 1] = (__bf16)v.y;
        out[i + 2] = (__bf16)v.z; out[i + 3] = (__bf16)v.w;
    }
}

// ---------------------------------------------------------------------------
// Tiled transpose + fp32->bf16: in[e][R][C] -> out[e][C][R].
// Produces K-contiguous weight layout so WMMA B-fragments are b128 loads.
// ---------------------------------------------------------------------------
__global__ __launch_bounds__(256) void transpose_cvt_kernel(
    const float* __restrict__ in, __bf16* __restrict__ out, int R, int C)
{
    __shared__ float tile[32][33];
    int e = blockIdx.z;
    const float* In  = in  + (size_t)e * R * C;
    __bf16*      Out = out + (size_t)e * R * C;
    int c0 = blockIdx.x * 32, r0 = blockIdx.y * 32;
#pragma unroll
    for (int i = 0; i < 32; i += 8)
        tile[threadIdx.y + i][threadIdx.x] =
            In[(size_t)(r0 + threadIdx.y + i) * C + (c0 + threadIdx.x)];
    __syncthreads();
#pragma unroll
    for (int i = 0; i < 32; i += 8)
        Out[(size_t)(c0 + threadIdx.y + i) * R + (r0 + threadIdx.x)] =
            (__bf16)tile[threadIdx.x][threadIdx.y + i];
}

// ---------------------------------------------------------------------------
// Main fused MoE kernel. One WG per 16-token tile, 8 waves.
// Dense over all experts (reference semantics), WMMA bf16, f32 accumulate.
// 4-way output-tile register blocking: one A fragment feeds 4 WMMAs
// (0.5 ds_load_b128 per WMMA), 4 independent accumulation chains.
// Dynamic LDS: xs[16][1024] bf16 (32KB) + hs[16][4096] bf16 (128KB) = 160KB.
// ---------------------------------------------------------------------------
#define NB 4   // output column tiles per register block

__global__ __launch_bounds__(256) void moe_main_kernel(
    const __bf16* __restrict__ xb,   // [N][D] bf16
    const __bf16* __restrict__ w1t,  // [E][H][D] bf16 (K-contiguous)
    const __bf16* __restrict__ w2t,  // [E][D][H] bf16 (K-contiguous)
    const float*  __restrict__ b1,   // [E][H]
    const float*  __restrict__ b2,   // [E][D]
    const int*    __restrict__ topi, // [N][2]
    const float*  __restrict__ topg, // [N][2]
    float*        __restrict__ out)  // [N][D]
{
    extern __shared__ char smem[];
    __bf16* xs = (__bf16*)smem;                              // [16][D_]
    __bf16* hs = (__bf16*)(smem + 16 * D_ * sizeof(__bf16)); // [16][H_]
    __shared__ float wgt[16][E_];

    const int tile = blockIdx.x;
    const int tid  = threadIdx.x;
    const int wave = tid >> 5;
    const int lane = tid & 31;
    const int half = lane >> 4;        // 0: lanes 0-15, 1: lanes 16-31
    const int lr   = lane & 15;        // row (A) / col (B,C,D) within tile
    const int klo  = half * 8;

    // Stage x tile into LDS (16*1024 bf16), 8B per thread per iter.
    const __bf16* xg = xb + (size_t)tile * 16 * D_;
    for (int i = tid * 4; i < 16 * D_; i += 256 * 4)
        *(float2*)(xs + i) = *(const float2*)(xg + i);

    // Per-token dense gate-weight row: w[t][e] = top-k gathered softmax weight.
    if (tid < 16) {
        int n = tile * 16 + tid;
#pragma unroll
        for (int e = 0; e < E_; ++e) wgt[tid][e] = 0.f;
        wgt[tid][topi[2 * n + 0]] = topg[2 * n + 0];
        wgt[tid][topi[2 * n + 1]] = topg[2 * n + 1];
    }
    __syncthreads();

    v8f oacc[8];
#pragma unroll
    for (int j = 0; j < 8; ++j)
#pragma unroll
        for (int r = 0; r < 8; ++r) oacc[j][r] = 0.f;

    for (int e = 0; e < E_; ++e) {
        float wrow[8];
#pragma unroll
        for (int r = 0; r < 8; ++r) wrow[r] = wgt[half * 8 + r][e];

        const __bf16* W1e = w1t + (size_t)e * H_ * D_;
        const __bf16* W2e = w2t + (size_t)e * D_ * H_;
        const float*  b1e = b1 + e * H_;
        const float*  b2e = b2 + e * D_;

        // ---- Layer 1: h[16][H] = relu(x @ W1[e] + b1[e]) ----
        // Each wave owns 32 column tiles; process NB at a time sharing A.
        for (int jb = 0; jb < 32; jb += NB) {
            int j0 = wave * 32 + jb;
            v8f acc[NB];
#pragma unroll
            for (int u = 0; u < NB; ++u) {
                float bias = b1e[(j0 + u) * 16 + lr];
#pragma unroll
                for (int r = 0; r < 8; ++r) acc[u][r] = bias;
            }
            for (int k0 = 0; k0 < D_; k0 += 32) {
                Frag a;
                load_frag(a, xs + lr * D_ + k0 + klo);
#pragma unroll
                for (int u = 0; u < NB; ++u) {
                    Frag b;
                    load_frag(b, W1e + (size_t)((j0 + u) * 16 + lr) * D_ + k0 + klo);
                    acc[u] = __builtin_amdgcn_wmma_f32_16x16x32_bf16(
                        false, a.v, false, b.v, (short)0, acc[u], false, false);
                }
            }
#pragma unroll
            for (int u = 0; u < NB; ++u)
#pragma unroll
                for (int r = 0; r < 8; ++r) {
                    float v = fmaxf(acc[u][r], 0.f);
                    hs[(half * 8 + r) * H_ + (j0 + u) * 16 + lr] = (__bf16)v;
                }
        }
        __syncthreads();

        // ---- Layer 2: out_e[16][D] = h @ W2[e] + b2[e]; scale & accumulate ----
        // Each wave owns 8 column tiles; process NB at a time sharing A.
        for (int jb = 0; jb < 8; jb += NB) {
            int j0 = wave * 8 + jb;
            v8f acc[NB];
#pragma unroll
            for (int u = 0; u < NB; ++u) {
                float bias = b2e[(j0 + u) * 16 + lr];
#pragma unroll
                for (int r = 0; r < 8; ++r) acc[u][r] = bias;
            }
            for (int k0 = 0; k0 < H_; k0 += 32) {
                Frag a;
                load_frag(a, hs + lr * H_ + k0 + klo);
#pragma unroll
                for (int u = 0; u < NB; ++u) {
                    Frag b;
                    load_frag(b, W2e + (size_t)((j0 + u) * 16 + lr) * H_ + k0 + klo);
                    acc[u] = __builtin_amdgcn_wmma_f32_16x16x32_bf16(
                        false, a.v, false, b.v, (short)0, acc[u], false, false);
                }
            }
#pragma unroll
            for (int u = 0; u < NB; ++u)
#pragma unroll
                for (int r = 0; r < 8; ++r)
                    oacc[jb + u][r] += wrow[r] * acc[u][r];
        }
        __syncthreads();  // hs is overwritten by next expert
    }

    // Store output tile (C/D layout: VGPR r -> row r+8*half, col lane&15).
#pragma unroll
    for (int jj = 0; jj < 8; ++jj) {
        int col = (wave * 8 + jj) * 16 + lr;
#pragma unroll
        for (int r = 0; r < 8; ++r)
            out[((size_t)tile * 16 + half * 8 + r) * D_ + col] = oacc[jj][r];
    }
}

// ---------------------------------------------------------------------------
extern "C" void kernel_launch(void* const* d_in, const int* in_sizes, int n_in,
                              void* d_out, int out_size, void* d_ws, size_t ws_size,
                              hipStream_t stream) {
    const float* x  = (const float*)d_in[0];
    const float* W1 = (const float*)d_in[1];
    const float* b1 = (const float*)d_in[2];
    const float* W2 = (const float*)d_in[3];
    const float* b2 = (const float*)d_in[4];
    const float* Wg = (const float*)d_in[5];
    const float* bg = (const float*)d_in[6];
    float* out = (float*)d_out;

    const int N = in_sizes[0] / D_;   // 8192 tokens

    // Workspace layout (~144.2 MB total)
    char* ws = (char*)d_ws;
    size_t off = 0;
    __bf16* xb  = (__bf16*)(ws + off); off += (size_t)N * D_ * sizeof(__bf16);
    __bf16* w1t = (__bf16*)(ws + off); off += (size_t)E_ * H_ * D_ * sizeof(__bf16);
    __bf16* w2t = (__bf16*)(ws + off); off += (size_t)E_ * D_ * H_ * sizeof(__bf16);
    int*   topi = (int*)(ws + off);    off += (size_t)N * 2 * sizeof(int);
    float* topg = (float*)(ws + off);  off += (size_t)N * 2 * sizeof(float);
    float* imp  = (float*)(ws + off);  off += E_ * sizeof(float);

    zero_imp_kernel<<<1, 32, 0, stream>>>(imp);
    gate_kernel<<<N / 4, 128, 0, stream>>>(x, Wg, bg, topi, topg, imp);
    cvt_bf16_kernel<<<(int)(((size_t)N * D_) / 1024), 256, 0, stream>>>(
        x, xb, (size_t)N * D_);
    dim3 tb(32, 8, 1);
    transpose_cvt_kernel<<<dim3(H_ / 32, D_ / 32, E_), tb, 0, stream>>>(W1, w1t, D_, H_);
    transpose_cvt_kernel<<<dim3(D_ / 32, H_ / 32, E_), tb, 0, stream>>>(W2, w2t, H_, D_);

    const size_t smem = (size_t)(16 * D_ + 16 * H_) * sizeof(__bf16); // 160 KB
    hipFuncSetAttribute((const void*)moe_main_kernel,
                        hipFuncAttributeMaxDynamicSharedMemorySize, (int)smem);
    moe_main_kernel<<<N / 16, 256, smem, stream>>>(xb, w1t, w2t, b1, b2,
                                                   topi, topg, out);
    aux_kernel<<<1, 1, 0, stream>>>(imp, out + (size_t)N * D_);
}
